// decode_GCN_46926812676545
// MI455X (gfx1250) — matrix operand order repaired and checked by reading, not compiled, not run on hardware
//
#include <hip/hip_runtime.h>
#include <math.h>

// ---------------------------------------------------------------------------
// out = ALPHA * softmax_col(M .* (LN(X@Qw^T+Qb) @ LN(H@Kw^T+Kb)^T * 0.125)) @ BN(H@Vw^T+Vb)  +  M @ H
//
// MI455X strategy (bandwidth-bound: M is 256MB f32, read ONCE):
//   P   = exp(M .* (Q@K^T)*0.125)  stored bf16 (128MB -> L2-resident)
//   csum[j] = sum_i P[i,j]         (column softmax denominator, f32 atomics)
//   V'  = ALPHA * V / csum         (bf16, stored TRANSPOSED for B-operand feeds)
//   out = P @ V' + M @ H           (M@H fused into the score pass)
// All big GEMMs on v_wmma_f32_16x16x32_bf16 (wave32, f32 accumulate).
// H and V' are kept transposed in global so every WMMA B operand is a
// contiguous 32B LDS read (ds_load_b128 x2) instead of 16 strided b32 reads.
// Tile staging uses GLOBAL_LOAD_ASYNC_TO_LDS_B128 (ASYNCcnt) when available.
// ---------------------------------------------------------------------------

typedef __attribute__((ext_vector_type(16))) __bf16        v16bf;
typedef __attribute__((ext_vector_type(8)))  float         v8f;
typedef __attribute__((ext_vector_type(4)))  unsigned int  u32x4;
typedef int v4i_g __attribute__((vector_size(16)));        // matches builtin param

union V16 { v16bf v; unsigned short u[16]; u32x4 q[2]; };
union V8F { v8f v;  float f[8]; };

__device__ __forceinline__ unsigned short f2bf(float f) {
    union { float f; unsigned u; } c; c.f = f;
    unsigned u = c.u + 0x7FFFu + ((c.u >> 16) & 1u);   // round-to-nearest-even
    return (unsigned short)(u >> 16);
}

#if defined(__gfx1250__) && __has_builtin(__builtin_amdgcn_global_load_async_to_lds_b128) && __has_builtin(__builtin_amdgcn_s_wait_asynccnt)
#define ASYNC_LDS 1
#endif

// 16B global -> LDS tile copy: async DMA path (ASYNCcnt, no VGPR roundtrip).
__device__ __forceinline__ void cp16(void* lds_dst, const void* gsrc) {
#ifdef ASYNC_LDS
    __builtin_amdgcn_global_load_async_to_lds_b128(
        (__attribute__((address_space(1))) v4i_g*)gsrc,
        (__attribute__((address_space(3))) v4i_g*)lds_dst, 0, 0);
#else
    *(u32x4*)lds_dst = *(const u32x4*)gsrc;
#endif
}
__device__ __forceinline__ void cp_wait() {
#ifdef ASYNC_LDS
    __builtin_amdgcn_s_wait_asynccnt(0);
#endif
}

#define N_ROWS   8192
#define NH_COLS  8192
#define COLSPLIT 2048   // columns handled per blockIdx.y
#define JTILES   (COLSPLIT / 64)

// ---------------------------------------------------------------------------
// Kernel 1: Q = LN(X@Qw^T+Qb) -> bf16, K = LN(H@Kw^T+Kb) -> bf16,
//           V = BN(H@Vw^T+Vb) -> f32, H -> bf16 TRANSPOSED [64][8192].
// ---------------------------------------------------------------------------
__global__ __launch_bounds__(64)
void k_qkv(const float* __restrict__ X, const float* __restrict__ H,
           const float* __restrict__ Qw, const float* __restrict__ Qb,
           const float* __restrict__ Kw, const float* __restrict__ Kb,
           const float* __restrict__ Vw, const float* __restrict__ Vb,
           const float* __restrict__ g,  const float* __restrict__ be,
           const float* __restrict__ mn, const float* __restrict__ vr,
           unsigned short* __restrict__ Qh, unsigned short* __restrict__ Kh,
           unsigned short* __restrict__ HhT, float* __restrict__ Vf)
{
    const int r = blockIdx.x, t = threadIdx.x;
    __shared__ float xr[64], hr[64], tm[64];
    xr[t] = X[r * 64 + t];
    hr[t] = H[r * 64 + t];
    __syncthreads();

    float q = Qb[t], k = Kb[t], v = Vb[t];
#pragma unroll
    for (int i = 0; i < 64; ++i) {
        q = fmaf(xr[i], Qw[t * 64 + i], q);
        k = fmaf(hr[i], Kw[t * 64 + i], k);
        v = fmaf(hr[i], Vw[t * 64 + i], v);
    }

    // LayerNorm(q)
    tm[t] = q; __syncthreads();
    float s = 0.f, s2 = 0.f;
#pragma unroll
    for (int i = 0; i < 64; ++i) { s += tm[i]; s2 += tm[i] * tm[i]; }
    float mu = s * (1.f / 64.f);
    float qn = (q - mu) * rsqrtf(s2 * (1.f / 64.f) - mu * mu + 1e-5f);
    __syncthreads();

    // LayerNorm(k)
    tm[t] = k; __syncthreads();
    s = 0.f; s2 = 0.f;
#pragma unroll
    for (int i = 0; i < 64; ++i) { s += tm[i]; s2 += tm[i] * tm[i]; }
    mu = s * (1.f / 64.f);
    float kn = (k - mu) * rsqrtf(s2 * (1.f / 64.f) - mu * mu + 1e-5f);

    // BatchNorm1d(v), eval mode
    float vb = (v - mn[t]) * rsqrtf(vr[t] + 1e-5f) * g[t] + be[t];

    Qh[r * 64 + t]       = f2bf(qn);
    Kh[r * 64 + t]       = f2bf(kn);
    HhT[t * NH_COLS + r] = f2bf(hr[t]);   // transposed: feature-major
    Vf[r * 64 + t]       = vb;
}

// ---------------------------------------------------------------------------
// Kernel 2: score pass. Block = 128 thr (4 waves) owns 64 output rows,
// walks a 2048-col split in 64-col tiles:
//   S = Qstrip @ Ktile^T (WMMA), P = exp(M.*S*0.125) -> bf16 global,
//   column sums -> LDS -> f32 atomics, MH += Mtile(bf16) @ Htile (WMMA).
// ---------------------------------------------------------------------------
__global__ __launch_bounds__(128)
void k_scores(const float* __restrict__ M,
              const unsigned short* __restrict__ Qh,
              const unsigned short* __restrict__ Kh,
              const unsigned short* __restrict__ HhT,
              unsigned short* __restrict__ P,
              float* __restrict__ csum,
              float* __restrict__ out)
{
    __shared__ __align__(16) float          lds_m[64 * 64];
    __shared__ __align__(16) unsigned short lds_k[64 * 64];   // [n][k] row-major
    __shared__ __align__(16) unsigned short lds_ht[64 * 64];  // [d][k] (transposed H)
    __shared__ __align__(16) unsigned short lds_q[64 * 64];
    __shared__ float lds_cs[64];

    const int t    = threadIdx.x;
    const int w    = t >> 5;        // wave id: 16-row strip
    const int lane = t & 31;
    const int row0 = blockIdx.x * 64;
    const int c0   = blockIdx.y * COLSPLIT;

    // Stage the block's 64x64 Q strip (contiguous) into LDS
#pragma unroll
    for (int i = 0; i < 4; ++i) {
        int idx = t + i * 128;
        cp16(&lds_q[idx * 8], Qh + (size_t)row0 * 64 + idx * 8);
    }
    cp_wait();
    __syncthreads();

    // Loop-invariant A operands (16x32 bf16 layout):
    // lane<16: rows 0..15, k' in {0..7,16..23}; lane>=16: k' in {8..15,24..31}
    const int arow = lane & 15;
    const int akb  = (lane >> 4) * 8;
    V16 qa[2];
#pragma unroll
    for (int ks = 0; ks < 2; ++ks) {
        const unsigned short* qr = &lds_q[(w * 16 + arow) * 64 + ks * 32 + akb];
#pragma unroll
        for (int i = 0; i < 8; ++i) { qa[ks].u[i] = qr[i]; qa[ks].u[8 + i] = qr[16 + i]; }
    }

    V8F mh[4];
#pragma unroll
    for (int d = 0; d < 4; ++d)
#pragma unroll
        for (int i = 0; i < 8; ++i) mh[d].f[i] = 0.f;

    const int bc = lane & 15;            // B/C column within a 16-wide tile
    const int bk = (lane >> 4) * 16;     // B K-half per lane group

    for (int jt = 0; jt < JTILES; ++jt) {
        const int j0 = c0 + jt * 64;

        // ---- stage tiles: M (f32, strided rows), K (contig), H^T (strided rows)
#pragma unroll
        for (int i = 0; i < 8; ++i) {
            int idx = t + i * 128, row = idx >> 4, c4 = idx & 15;
            cp16(&lds_m[idx * 4], M + (size_t)(row0 + row) * NH_COLS + j0 + c4 * 4);
        }
#pragma unroll
        for (int i = 0; i < 4; ++i) {
            int idx = t + i * 128, row = idx >> 3, c8 = idx & 7;
            cp16(&lds_k[idx * 8], Kh + (size_t)j0 * 64 + idx * 8);
            cp16(&lds_ht[row * 64 + c8 * 8],
                 HhT + (size_t)row * NH_COLS + j0 + c8 * 8);
        }
        if (t < 64) lds_cs[t] = 0.f;
        if (jt + 1 < JTILES)   // speculative prefetch of next M tile rows
            __builtin_prefetch(M + (size_t)(row0 + (t >> 1)) * NH_COLS + j0 + 64, 0, 1);
        cp_wait();
        __syncthreads();

        // A operand for M@H: wave's 16x64 M strip -> bf16, A layout
        V16 ma[2];
#pragma unroll
        for (int ks = 0; ks < 2; ++ks) {
            const float* mr = &lds_m[(w * 16 + arow) * 64 + ks * 32 + akb];
#pragma unroll
            for (int i = 0; i < 8; ++i) { ma[ks].u[i] = f2bf(mr[i]); ma[ks].u[8 + i] = f2bf(mr[16 + i]); }
        }

        // ---- scores: 8 independent WMMAs first (pipelines, hides hazards)
        V8F acc[4];
#pragma unroll
        for (int tile = 0; tile < 4; ++tile) {
            V16 b0, b1;    // B[k',n] = Krow(n)[k'] : contiguous per lane
            const unsigned short* kr = &lds_k[(tile * 16 + bc) * 64 + akb];
            b0.q[0] = ((const u32x4*)kr)[0];
            b0.q[1] = ((const u32x4*)(kr + 16))[0];
            b1.q[0] = ((const u32x4*)(kr + 32))[0];
            b1.q[1] = ((const u32x4*)(kr + 48))[0];
#pragma unroll
            for (int i = 0; i < 8; ++i) acc[tile].f[i] = 0.f;
            acc[tile].v = __builtin_amdgcn_wmma_f32_16x16x32_bf16(
                false, qa[0].v, false, b0.v, (short)0, acc[tile].v, false, false);
            acc[tile].v = __builtin_amdgcn_wmma_f32_16x16x32_bf16(
                false, qa[1].v, false, b1.v, (short)0, acc[tile].v, false, false);
        }

        // ---- epilogue: mask*0.125, exp, emit P bf16, column sums
#pragma unroll
        for (int tile = 0; tile < 4; ++tile) {
            float cs = 0.f;
#pragma unroll
            for (int r = 0; r < 8; ++r) {
                int m  = (lane >> 4) * 8 + r;          // C layout: lane16 split
                int cl = tile * 16 + bc;
                float mv = lds_m[(w * 16 + m) * 64 + cl];
                float p  = __expf(mv * acc[tile].f[r] * 0.125f);
                cs += p;
                P[(size_t)(row0 + w * 16 + m) * NH_COLS + j0 + cl] = f2bf(p);
            }
            cs += __shfl_xor(cs, 16, 32);             // join row halves of column
            if (lane < 16) atomicAdd(&lds_cs[tile * 16 + lane], cs);
        }

        // ---- fused M@H: B[k',d] = H^T[d][k'] : contiguous per lane
#pragma unroll
        for (int tile = 0; tile < 4; ++tile)
#pragma unroll
            for (int ks = 0; ks < 2; ++ks) {
                V16 hb;
                const unsigned short* hp = &lds_ht[(tile * 16 + bc) * 64 + ks * 32 + akb];
                hb.q[0] = ((const u32x4*)hp)[0];
                hb.q[1] = ((const u32x4*)(hp + 16))[0];
                mh[tile].v = __builtin_amdgcn_wmma_f32_16x16x32_bf16(
                    false, ma[ks].v, false, hb.v, (short)0, mh[tile].v, false, false);
            }

        __syncthreads();
        if (t < 64) atomicAdd(&csum[j0 + t], lds_cs[t]);
        __syncthreads();
    }

    // out = M@H partial (col-splits combine via f32 atomics; out pre-zeroed)
#pragma unroll
    for (int tile = 0; tile < 4; ++tile)
#pragma unroll
        for (int r = 0; r < 8; ++r) {
            int m = (lane >> 4) * 8 + r;
            atomicAdd(&out[(size_t)(row0 + w * 16 + m) * 64 + tile * 16 + (lane & 15)],
                      mh[tile].f[r]);
        }
}

// ---------------------------------------------------------------------------
// Kernel 3: V'^T[d][j] = ALPHA * V[j][d] / csum[j]  -> bf16 (transposed)
// ---------------------------------------------------------------------------
__global__ __launch_bounds__(256)
void k_vscale(const float* __restrict__ Vf, const float* __restrict__ csum,
              unsigned short* __restrict__ VsT)
{
    int idx = blockIdx.x * 256 + threadIdx.x;     // 8192*64 elements, j-major
    int j = idx >> 6, d = idx & 63;
    VsT[(size_t)d * NH_COLS + j] = f2bf(0.1f * Vf[idx] / csum[j]);
}

// ---------------------------------------------------------------------------
// Kernel 4: out += P @ V'   (P bf16 streamed from L2; V'^T tile in LDS)
// ---------------------------------------------------------------------------
__global__ __launch_bounds__(128)
void k_pav(const unsigned short* __restrict__ P,
           const unsigned short* __restrict__ VsT,
           float* __restrict__ out)
{
    __shared__ __align__(16) unsigned short lds_vt[64 * 64];  // [d][k]

    const int t    = threadIdx.x;
    const int w    = t >> 5;
    const int lane = t & 31;
    const int row0 = blockIdx.x * 64;
    const int c0   = blockIdx.y * COLSPLIT;

    const int arow = lane & 15;
    const int akb  = (lane >> 4) * 8;
    const int bc   = lane & 15;
    const int bk   = (lane >> 4) * 16;

    V8F acc[4];
#pragma unroll
    for (int d = 0; d < 4; ++d)
#pragma unroll
        for (int i = 0; i < 8; ++i) acc[d].f[i] = 0.f;

    for (int jt = 0; jt < JTILES; ++jt) {
        const int j0 = c0 + jt * 64;
#pragma unroll
        for (int i = 0; i < 4; ++i) {
            int idx = t + i * 128, row = idx >> 3, c8 = idx & 7;
            cp16(&lds_vt[row * 64 + c8 * 8],
                 VsT + (size_t)row * NH_COLS + j0 + c8 * 8);
        }
        cp_wait();
        __syncthreads();

        // A operand: 16x64 strip of P (bf16), two 16B global loads per k-step
        const unsigned short* pr =
            P + (size_t)(row0 + w * 16 + arow) * NH_COLS + j0;
        V16 pa[2];
#pragma unroll
        for (int ks = 0; ks < 2; ++ks) {
            pa[ks].q[0] = *(const u32x4*)(pr + ks * 32 + akb);
            pa[ks].q[1] = *(const u32x4*)(pr + ks * 32 + akb + 16);
        }

#pragma unroll
        for (int tile = 0; tile < 4; ++tile)
#pragma unroll
            for (int ks = 0; ks < 2; ++ks) {
                V16 vb;   // B[k',d] = V'^T[d][k'] : contiguous per lane
                const unsigned short* vp = &lds_vt[(tile * 16 + bc) * 64 + ks * 32 + akb];
                vb.q[0] = ((const u32x4*)vp)[0];
                vb.q[1] = ((const u32x4*)(vp + 16))[0];
                acc[tile].v = __builtin_amdgcn_wmma_f32_16x16x32_bf16(
                    false, pa[ks].v, false, vb.v, (short)0, acc[tile].v, false, false);
            }
        __syncthreads();
    }

#pragma unroll
    for (int tile = 0; tile < 4; ++tile)
#pragma unroll
        for (int r = 0; r < 8; ++r) {
            int m = (lane >> 4) * 8 + r;
            atomicAdd(&out[(size_t)(row0 + w * 16 + m) * 64 + tile * 16 + (lane & 15)],
                      acc[tile].f[r]);
        }
}

// ---------------------------------------------------------------------------
extern "C" void kernel_launch(void* const* d_in, const int* in_sizes, int n_in,
                              void* d_out, int out_size, void* d_ws, size_t ws_size,
                              hipStream_t stream)
{
    const float* X  = (const float*)d_in[0];
    const float* H  = (const float*)d_in[1];
    const float* M  = (const float*)d_in[2];
    const float* Qw = (const float*)d_in[3];
    const float* Qb = (const float*)d_in[4];
    const float* Kw = (const float*)d_in[5];
    const float* Kb = (const float*)d_in[6];
    const float* Vw = (const float*)d_in[7];
    const float* Vb = (const float*)d_in[8];
    const float* g  = (const float*)d_in[9];
    const float* be = (const float*)d_in[10];
    const float* mn = (const float*)d_in[11];
    const float* vr = (const float*)d_in[12];
    float* out = (float*)d_out;

    const size_t MB = 1024 * 1024;
    char* w = (char*)d_ws;
    unsigned short* Qh   = (unsigned short*)(w + 0 * MB);  // 1 MB
    unsigned short* Kh   = (unsigned short*)(w + 1 * MB);  // 1 MB
    unsigned short* HhT  = (unsigned short*)(w + 2 * MB);  // 1 MB (transposed)
    float*          Vf   = (float*)         (w + 3 * MB);  // 2 MB
    unsigned short* VsT  = (unsigned short*)(w + 5 * MB);  // 1 MB (transposed)
    float*          csum = (float*)         (w + 6 * MB);  // 32 KB
    unsigned short* P    = (unsigned short*)(w + 8 * MB);  // 128 MB

    (void)hipMemsetAsync(out,  0, (size_t)N_ROWS * 64 * sizeof(float), stream);
    (void)hipMemsetAsync(csum, 0, (size_t)NH_COLS * sizeof(float), stream);

    k_qkv<<<N_ROWS, 64, 0, stream>>>(X, H, Qw, Qb, Kw, Kb, Vw, Vb,
                                     g, be, mn, vr, Qh, Kh, HhT, Vf);
    k_scores<<<dim3(N_ROWS / 64, NH_COLS / COLSPLIT), 128, 0, stream>>>(
        M, Qh, Kh, HhT, P, csum, out);
    k_vscale<<<(NH_COLS * 64) / 256, 256, 0, stream>>>(Vf, csum, VsT);
    k_pav<<<dim3(N_ROWS / 64, NH_COLS / COLSPLIT), 128, 0, stream>>>(P, VsT, out);
}
